// Generator_26285199851535
// MI455X (gfx1250) — matrix operand order repaired and checked by reading, not compiled, not run on hardware
//
#include <hip/hip_runtime.h>

#define BS 8
#define D 512
#define C 128
#define L_MAP 8
#define N_LAYERS 9
#define IMG 64

typedef __attribute__((ext_vector_type(16))) _Float16 v16h;
typedef __attribute__((ext_vector_type(8)))  _Float16 v8h;
typedef __attribute__((ext_vector_type(8)))  float    v8f;

__device__ __forceinline__ float leaky_f(float v) { return v >= 0.f ? v : 0.2f * v; }

__device__ __forceinline__ v16h cat16(v8h lo, v8h hi) {
    return __builtin_shufflevector(lo, hi, 0,1,2,3,4,5,6,7,8,9,10,11,12,13,14,15);
}

// ---------------------------------------------------------------------------
// 1) Convert conv weights fp32 [i][co][ci][kh][kw] -> f16 [i][k][co][ci]
// ---------------------------------------------------------------------------
__global__ __launch_bounds__(256)
void prep_w16_kernel(const float* __restrict__ conv_w, _Float16* __restrict__ w16) {
    int idx = blockIdx.x * 256 + threadIdx.x;
    const int total = N_LAYERS * 9 * C * C;
    if (idx >= total) return;
    int ci = idx & (C - 1);
    int r  = idx >> 7;
    int co = r & (C - 1); r >>= 7;
    int k  = r % 9;
    int i  = r / 9;
    float v = conv_w[(((size_t)(i * C + co) * C + ci) * 9) + k];
    w16[idx] = (_Float16)v;
}

// ---------------------------------------------------------------------------
// 2) Mapping network: pixelnorm + 8x (FC 512x512 + leaky). One block, 512 thr.
// ---------------------------------------------------------------------------
__global__ __launch_bounds__(512)
void mapping_kernel(const float* __restrict__ z, const float* __restrict__ map_w,
                    const float* __restrict__ map_b, float* __restrict__ w_out) {
    __shared__ float wl[BS * D];
    __shared__ float scale[BS];
    const int tid = threadIdx.x;

    if (tid < BS) {
        float s = 0.f;
        for (int d = 0; d < D; ++d) { float v = z[tid * D + d]; s += v * v; }
        scale[tid] = rsqrtf(s / (float)D + 1e-8f);
    }
    __syncthreads();
    for (int b = 0; b < BS; ++b) wl[b * D + tid] = z[b * D + tid] * scale[b];
    __syncthreads();

    for (int l = 0; l < L_MAP; ++l) {
        float acc[BS];
        float bias = map_b[l * D + tid];
        #pragma unroll
        for (int b = 0; b < BS; ++b) acc[b] = bias;
        for (int d = 0; d < D; ++d) {
            float mw = map_w[((size_t)(l * D + d)) * D + tid];
            #pragma unroll
            for (int b = 0; b < BS; ++b) acc[b] += wl[b * D + d] * mw;
        }
        __syncthreads();
        #pragma unroll
        for (int b = 0; b < BS; ++b) wl[b * D + tid] = leaky_f(acc[b]);
        __syncthreads();
    }
    for (int b = 0; b < BS; ++b) w_out[b * D + tid] = wl[b * D + tid];
}

// ---------------------------------------------------------------------------
// 3) Per-layer style + demod factor
// ---------------------------------------------------------------------------
__global__ __launch_bounds__(C)
void style_demod_kernel(const float* __restrict__ w, const float* __restrict__ A_w,
                        const float* __restrict__ A_b, const float* __restrict__ conv_w,
                        float* __restrict__ styles, float* __restrict__ rstd) {
    const int i = blockIdx.x;   // layer
    const int c = threadIdx.x;  // channel
    __shared__ float s2[BS * C];

    float st[BS];
    float bias = A_b[i * C + c];
    #pragma unroll
    for (int b = 0; b < BS; ++b) st[b] = bias;
    for (int d = 0; d < D; ++d) {
        float aw = A_w[((size_t)(i * D + d)) * C + c];
        #pragma unroll
        for (int b = 0; b < BS; ++b) st[b] += w[b * D + d] * aw;
    }
    #pragma unroll
    for (int b = 0; b < BS; ++b) {
        styles[((size_t)i * BS + b) * C + c] = st[b];
        s2[b * C + c] = st[b] * st[b];
    }
    __syncthreads();

    float acc[BS];
    #pragma unroll
    for (int b = 0; b < BS; ++b) acc[b] = 0.f;
    const float* wl = conv_w + (size_t)i * C * C * 9;
    for (int ci = 0; ci < C; ++ci) {
        const float* wp = wl + ((size_t)c * C + ci) * 9;
        float wsq = 0.f;
        #pragma unroll
        for (int k = 0; k < 9; ++k) { float v = wp[k]; wsq += v * v; }
        #pragma unroll
        for (int b = 0; b < BS; ++b) acc[b] += s2[b * C + ci] * wsq;
    }
    #pragma unroll
    for (int b = 0; b < BS; ++b)
        rstd[((size_t)i * BS + b) * C + c] = rsqrtf(acc[b] + 1e-8f);
}

// ---------------------------------------------------------------------------
// 4) Zero-init
// ---------------------------------------------------------------------------
__global__ __launch_bounds__(256)
void zero_kernel(float* __restrict__ p, int n) {
    int idx = blockIdx.x * 256 + threadIdx.x;
    if (idx < n) p[idx] = 0.f;
}

// ---------------------------------------------------------------------------
// 5a) Pre-modulation (same resolution): xs[b][y+1][x+1][ci] = f16(x * style)
// ---------------------------------------------------------------------------
__global__ __launch_bounds__(256)
void premod_kernel(const float* __restrict__ x, const float* __restrict__ style,
                   _Float16* __restrict__ xs, int H, int W) {
    const int Hp = H + 2, Wp = W + 2;
    int idx = blockIdx.x * 256 + threadIdx.x;
    int total = BS * Hp * Wp * C;
    if (idx >= total) return;
    int ci = idx & (C - 1); int r = idx >> 7;
    int sx = r % Wp; r /= Wp;
    int sy = r % Hp; int b = r / Hp;
    float v = 0.f;
    if (sy >= 1 && sy <= H && sx >= 1 && sx <= W)
        v = x[(((size_t)b * C + ci) * H + (sy - 1)) * W + (sx - 1)] * style[b * C + ci];
    xs[idx] = (_Float16)v;
}

// ---------------------------------------------------------------------------
// 5b) Fused bilinear-2x-upsample + pre-modulation:
//     reads x at (H,W), writes f16 halo tensor at (2H,2W). Saves a full fp32
//     round-trip of the activation tensor per odd layer.
// ---------------------------------------------------------------------------
__global__ __launch_bounds__(256)
void premod_up_kernel(const float* __restrict__ x, const float* __restrict__ style,
                      _Float16* __restrict__ xs, int H, int W) {
    const int OH = 2 * H, OW = 2 * W;
    const int Hp = OH + 2, Wp = OW + 2;
    int idx = blockIdx.x * 256 + threadIdx.x;
    int total = BS * Hp * Wp * C;
    if (idx >= total) return;
    int ci = idx & (C - 1); int r = idx >> 7;
    int sx = r % Wp; r /= Wp;
    int sy = r % Hp; int b = r / Hp;
    float v = 0.f;
    if (sy >= 1 && sy <= OH && sx >= 1 && sx <= OW) {
        int oy = sy - 1, ox = sx - 1;
        float fsy = oy * 0.5f - 0.25f;
        float fsx = ox * 0.5f - 0.25f;
        int y0 = (int)floorf(fsy), x0 = (int)floorf(fsx);
        float fy = fsy - (float)y0, fx = fsx - (float)x0;
        int y0c = y0 < 0 ? 0 : (y0 >= H ? H - 1 : y0);
        int y1c = (y0 + 1) >= H ? H - 1 : (y0 + 1 < 0 ? 0 : y0 + 1);
        int x0c = x0 < 0 ? 0 : (x0 >= W ? W - 1 : x0);
        int x1c = (x0 + 1) >= W ? W - 1 : (x0 + 1 < 0 ? 0 : x0 + 1);
        const float* p = x + ((size_t)b * C + ci) * H * W;
        float v00 = p[y0c * W + x0c], v01 = p[y0c * W + x1c];
        float v10 = p[y1c * W + x0c], v11 = p[y1c * W + x1c];
        v = (1.f - fy) * ((1.f - fx) * v00 + fx * v01)
          +        fy  * ((1.f - fx) * v10 + fx * v11);
        v *= style[b * C + ci];
    }
    xs[idx] = (_Float16)v;
}

// ---------------------------------------------------------------------------
// 6a) WMMA conv, 4x4 pixel tile (layer 0 only, H=W=4). One 16x16 tile / wave.
// ---------------------------------------------------------------------------
__global__ __launch_bounds__(256)
void conv_mod_wmma16_kernel(const _Float16* __restrict__ xs,
                            const _Float16* __restrict__ w16,
                            const float* __restrict__ rstd,
                            const float* __restrict__ nz,
                            const float* __restrict__ Bw, const float* __restrict__ Bb,
                            float* __restrict__ xout, int H, int W) {
    const int b = blockIdx.x;           // one 4x4 tile per sample
    const int tid  = threadIdx.x;
    const int wave = tid >> 5;
    const int lane = tid & 31;
    const int Wp = W + 2;

    __shared__ _Float16 lds[6 * 6 * C];
    for (int idx = tid; idx < 36 * 16; idx += 256) {
        int s = idx >> 4, c8 = (idx & 15) << 3;
        int sy = s / 6, sx = s % 6;
        size_t g = (((size_t)b * (H + 2) + sy) * Wp + sx) * C + c8;
        *(v8h*)&lds[s * C + c8] = *(const v8h*)(xs + g);
    }
    __syncthreads();

    const int co0  = wave << 4;
    const int rowA = lane & 15;
    const int kA   = (lane >> 4) << 3;
    const int nB   = lane & 15;
    const int py   = nB >> 2, px = nB & 3;
    const int kB   = (lane >> 4) << 3;

    v8f acc = {0.f, 0.f, 0.f, 0.f, 0.f, 0.f, 0.f, 0.f};
    for (int k = 0; k < 9; ++k) {
        const int kh = k / 3, kw = k % 3;
        const int sbase = ((py + kh) * 6 + (px + kw)) * C;
        const _Float16* wbase = w16 + ((size_t)k * C + (co0 + rowA)) * C;
        #pragma unroll
        for (int ci0 = 0; ci0 < C; ci0 += 32) {
            v16h a = cat16(*(const v8h*)(wbase + ci0 + kA),
                           *(const v8h*)(wbase + ci0 + kA + 16));
            v16h bm = cat16(*(const v8h*)(&lds[sbase + ci0 + kB]),
                            *(const v8h*)(&lds[sbase + ci0 + kB + 16]));
            acc = __builtin_amdgcn_wmma_f32_16x16x32_f16(
                false, a, false, bm, (short)0, acc, false, false);
        }
    }

    const float nv = nz[(size_t)b * IMG * IMG + py * W + px];
    const int mBase = (lane >> 4) << 3;
    #pragma unroll
    for (int v = 0; v < 8; ++v) {
        int co = co0 + v + mBase;
        float val = acc[v] * rstd[b * C + co] + nv * Bw[co] + Bb[co];
        xout[(((size_t)b * C + co) * H + py) * W + px] = leaky_f(val);
    }
}

// ---------------------------------------------------------------------------
// 6b) WMMA conv, 8x8 pixel tile, 2x2 register-blocked (layers 1..8, H>=8).
//     8 waves = 4 co-groups (32 co each) x 2 pixel-groups (2 subtiles each).
//     Each A/B fragment feeds 2 WMMAs -> 4 wmma per 8 b128 loads.
// ---------------------------------------------------------------------------
__global__ __launch_bounds__(256)
void conv_mod_wmma64_kernel(const _Float16* __restrict__ xs,
                            const _Float16* __restrict__ w16,
                            const float* __restrict__ rstd,
                            const float* __restrict__ nz,
                            const float* __restrict__ Bw, const float* __restrict__ Bb,
                            float* __restrict__ xout, int H, int W) {
    const int tilesX = W >> 3, tilesY = H >> 3;
    int bid = blockIdx.x;
    int b   = bid / (tilesX * tilesY);
    int t   = bid % (tilesX * tilesY);
    int ty  = t / tilesX, tx = t % tilesX;

    const int tid  = threadIdx.x;
    const int wave = tid >> 5;
    const int lane = tid & 31;
    const int Wp = W + 2;

    // 10x10 spatial x 128 ci f16 halo tile = 25600 B
    __shared__ _Float16 lds[10 * 10 * C];
    for (int idx = tid; idx < 100 * 16; idx += 256) {
        int s = idx >> 4, c8 = (idx & 15) << 3;
        int sy = s / 10, sx = s % 10;
        size_t g = (((size_t)b * (H + 2) + (ty * 8 + sy)) * Wp + (tx * 8 + sx)) * C + c8;
        *(v8h*)&lds[s * C + c8] = *(const v8h*)(xs + g);
    }
    __syncthreads();

    const int coGroup  = wave & 3;     // co0 = 32*coGroup, two 16-co tiles
    const int pixGroup = wave >> 2;    // subtiles {2*pg, 2*pg+1} of the 2x2 grid
    const int co0  = coGroup << 5;
    const int rowA = lane & 15;
    const int kA   = (lane >> 4) << 3;
    const int nB   = lane & 15;
    const int py   = nB >> 2, px = nB & 3;
    const int kB   = (lane >> 4) << 3;

    // subtile spatial origins within the 8x8 block tile
    int sy0[2], sx0[2];
    #pragma unroll
    for (int n = 0; n < 2; ++n) {
        int st = pixGroup * 2 + n;
        sy0[n] = (st >> 1) << 2;
        sx0[n] = (st & 1) << 2;
    }

    v8f zero = {0.f, 0.f, 0.f, 0.f, 0.f, 0.f, 0.f, 0.f};
    v8f acc00 = zero, acc01 = zero, acc10 = zero, acc11 = zero;

    for (int k = 0; k < 9; ++k) {
        const int kh = k / 3, kw = k % 3;
        const _Float16* wb0 = w16 + ((size_t)k * C + (co0 + rowA)) * C;
        const _Float16* wb1 = wb0 + (size_t)16 * C;
        __builtin_prefetch(wb0 + C, 0, 1);
        const int sb0 = ((sy0[0] + py + kh) * 10 + (sx0[0] + px + kw)) * C;
        const int sb1 = ((sy0[1] + py + kh) * 10 + (sx0[1] + px + kw)) * C;
        #pragma unroll
        for (int ci0 = 0; ci0 < C; ci0 += 32) {
            v16h a0 = cat16(*(const v8h*)(wb0 + ci0 + kA),
                            *(const v8h*)(wb0 + ci0 + kA + 16));
            v16h a1 = cat16(*(const v8h*)(wb1 + ci0 + kA),
                            *(const v8h*)(wb1 + ci0 + kA + 16));
            v16h b0 = cat16(*(const v8h*)(&lds[sb0 + ci0 + kB]),
                            *(const v8h*)(&lds[sb0 + ci0 + kB + 16]));
            v16h b1 = cat16(*(const v8h*)(&lds[sb1 + ci0 + kB]),
                            *(const v8h*)(&lds[sb1 + ci0 + kB + 16]));
            acc00 = __builtin_amdgcn_wmma_f32_16x16x32_f16(false, a0, false, b0, (short)0, acc00, false, false);
            acc01 = __builtin_amdgcn_wmma_f32_16x16x32_f16(false, a0, false, b1, (short)0, acc01, false, false);
            acc10 = __builtin_amdgcn_wmma_f32_16x16x32_f16(false, a1, false, b0, (short)0, acc10, false, false);
            acc11 = __builtin_amdgcn_wmma_f32_16x16x32_f16(false, a1, false, b1, (short)0, acc11, false, false);
        }
    }

    const int mBase = (lane >> 4) << 3;
    #pragma unroll
    for (int n = 0; n < 2; ++n) {
        int gy = ty * 8 + sy0[n] + py;
        int gx = tx * 8 + sx0[n] + px;
        float nv = nz[(size_t)b * IMG * IMG + gy * W + gx];
        #pragma unroll
        for (int m = 0; m < 2; ++m) {
            v8f accv = (m == 0) ? (n == 0 ? acc00 : acc01)
                                : (n == 0 ? acc10 : acc11);
            #pragma unroll
            for (int v = 0; v < 8; ++v) {
                int co = co0 + m * 16 + v + mBase;
                float val = accv[v] * rstd[b * C + co] + nv * Bw[co] + Bb[co];
                xout[(((size_t)b * C + co) * H + gy) * W + gx] = leaky_f(val);
            }
        }
    }
}

// ---------------------------------------------------------------------------
// 7) toRGB: 3-channel 3x3 conv + bias (tiny; VALU)
// ---------------------------------------------------------------------------
__global__ __launch_bounds__(256)
void to_rgb_kernel(const float* __restrict__ x, const float* __restrict__ rgb_w,
                   const float* __restrict__ rgb_b, float* __restrict__ out) {
    int idx = blockIdx.x * 256 + threadIdx.x;
    if (idx >= BS * IMG * IMG) return;
    int pix = idx & (IMG * IMG - 1);
    int b   = idx / (IMG * IMG);
    int py = pix >> 6, px = pix & 63;
    float acc0 = rgb_b[0], acc1 = rgb_b[1], acc2 = rgb_b[2];
    for (int ci = 0; ci < C; ++ci) {
        const float* xp = x + (((size_t)b * C + ci) * IMG) * IMG;
        #pragma unroll
        for (int kh = 0; kh < 3; ++kh) {
            int yy = py + kh - 1;
            if (yy < 0 || yy >= IMG) continue;
            #pragma unroll
            for (int kw = 0; kw < 3; ++kw) {
                int xx = px + kw - 1;
                if (xx < 0 || xx >= IMG) continue;
                float xv = xp[yy * IMG + xx];
                acc0 += xv * rgb_w[((0 * C + ci) * 3 + kh) * 3 + kw];
                acc1 += xv * rgb_w[((1 * C + ci) * 3 + kh) * 3 + kw];
                acc2 += xv * rgb_w[((2 * C + ci) * 3 + kh) * 3 + kw];
            }
        }
    }
    out[((size_t)(b * 3 + 0) * IMG + py) * IMG + px] = acc0;
    out[((size_t)(b * 3 + 1) * IMG + py) * IMG + px] = acc1;
    out[((size_t)(b * 3 + 2) * IMG + py) * IMG + px] = acc2;
}

// ---------------------------------------------------------------------------
// Host driver
// ---------------------------------------------------------------------------
extern "C" void kernel_launch(void* const* d_in, const int* in_sizes, int n_in,
                              void* d_out, int out_size, void* d_ws, size_t ws_size,
                              hipStream_t stream) {
    const float* latent_z = (const float*)d_in[0];
    const float* noise    = (const float*)d_in[1];
    const float* map_w    = (const float*)d_in[2];
    const float* map_b    = (const float*)d_in[3];
    const float* A_w      = (const float*)d_in[4];
    const float* A_b      = (const float*)d_in[5];
    const float* B_w      = (const float*)d_in[6];
    const float* B_b      = (const float*)d_in[7];
    const float* conv_w   = (const float*)d_in[8];
    const float* rgb_w    = (const float*)d_in[9];
    const float* rgb_b    = (const float*)d_in[10];
    float* out = (float*)d_out;

    char* base = (char*)d_ws;
    size_t off = 0;
    auto carve = [&](size_t bytes) -> void* {
        void* p = base + off;
        off += (bytes + 255) & ~(size_t)255;
        return p;
    };
    float*    w_map  = (float*)   carve((size_t)BS * D * 4);
    float*    styles = (float*)   carve((size_t)N_LAYERS * BS * C * 4);
    float*    rstd   = (float*)   carve((size_t)N_LAYERS * BS * C * 4);
    _Float16* w16    = (_Float16*)carve((size_t)N_LAYERS * 9 * C * C * 2);
    float*    xA     = (float*)   carve((size_t)BS * C * IMG * IMG * 4);
    float*    xB     = (float*)   carve((size_t)BS * C * IMG * IMG * 4);
    _Float16* xs     = (_Float16*)carve((size_t)BS * (IMG + 2) * (IMG + 2) * C * 2);

    {
        int total = N_LAYERS * 9 * C * C;
        prep_w16_kernel<<<(total + 255) / 256, 256, 0, stream>>>(conv_w, w16);
    }
    mapping_kernel<<<1, 512, 0, stream>>>(latent_z, map_w, map_b, w_map);
    style_demod_kernel<<<N_LAYERS, C, 0, stream>>>(w_map, A_w, A_b, conv_w, styles, rstd);

    zero_kernel<<<(BS * C * 16 + 255) / 256, 256, 0, stream>>>(xA, BS * C * 16);
    float* x = xA;
    float* t = xB;
    int H = 4, W = 4;
    for (int i = 0; i < N_LAYERS; ++i) {
        if (i & 1) {
            // fused bilinear 2x upsample + style modulation -> f16 halo tensor
            int OH = 2 * H, OW = 2 * W;
            int total = BS * (OH + 2) * (OW + 2) * C;
            premod_up_kernel<<<(total + 255) / 256, 256, 0, stream>>>(
                x, styles + (size_t)i * BS * C, xs, H, W);
            H = OH; W = OW;
        } else {
            int total = BS * (H + 2) * (W + 2) * C;
            premod_kernel<<<(total + 255) / 256, 256, 0, stream>>>(
                x, styles + (size_t)i * BS * C, xs, H, W);
        }
        if (H == 4) {
            conv_mod_wmma16_kernel<<<BS, 256, 0, stream>>>(
                xs, w16 + (size_t)i * 9 * C * C,
                rstd + (size_t)i * BS * C,
                noise + (size_t)i * BS * IMG * IMG,
                B_w + (size_t)i * C, B_b + (size_t)i * C,
                t, H, W);
        } else {
            int blocks = BS * (H / 8) * (W / 8);
            conv_mod_wmma64_kernel<<<blocks, 256, 0, stream>>>(
                xs, w16 + (size_t)i * 9 * C * C,
                rstd + (size_t)i * BS * C,
                noise + (size_t)i * BS * IMG * IMG,
                B_w + (size_t)i * C, B_b + (size_t)i * C,
                t, H, W);
        }
        float* s = x; x = t; t = s;
    }

    to_rgb_kernel<<<(BS * IMG * IMG + 255) / 256, 256, 0, stream>>>(x, rgb_w, rgb_b, out);
}
